// Model_STNv2_15728170238252
// MI455X (gfx1250) — matrix operand order repaired and checked by reading, not compiled, not run on hardware
//
#include <hip/hip_runtime.h>
#include <hip/hip_bf16.h>
#include <math.h>

typedef _Float16 v8h  __attribute__((ext_vector_type(8)));
typedef _Float16 v16h __attribute__((ext_vector_type(16)));
typedef float    v8f  __attribute__((ext_vector_type(8)));

#define BM 64
#define BN 64
#define BK 32
#define LDT 40   // padded LDS row stride (halfwords) -> conflict-free b128 reads

#if defined(__HIP_DEVICE_COMPILE__) && __has_builtin(__builtin_amdgcn_global_load_async_to_lds_b128)
#define HAVE_ASYNC_LDS 1
typedef int v4i __attribute__((vector_size(16)));
typedef __attribute__((address_space(1))) v4i g_v4i;   // clang prints AS1 as "__device__"
typedef __attribute__((address_space(3))) v4i l_v4i;   // LDS
#else
#define HAVE_ASYNC_LDS 0
#endif

__device__ __forceinline__ void wait_asynccnt0() {
#if defined(__HIP_DEVICE_COMPILE__)
#if __has_builtin(__builtin_amdgcn_s_wait_asynccnt)
  __builtin_amdgcn_s_wait_asynccnt(0);
#elif HAVE_ASYNC_LDS
  asm volatile("s_wait_asynccnt 0" ::: "memory");
#endif
#endif
}

// ---------------- init GEMM: pred0 = last_feature @ W_trans + b_trans ----------------
__global__ __launch_bounds__(128)
void k_pred0(const float* __restrict__ lf, const float* __restrict__ Wt,
             const float* __restrict__ bt, float* __restrict__ pred) {
  __shared__ float red[128];
  const int b = blockIdx.x, col = blockIdx.y, t = threadIdx.x;
  float s = 0.f;
  for (int k = t; k < 4096; k += 128)
    s += lf[b * 4096 + k] * Wt[k * 18 + col];
  red[t] = s; __syncthreads();
  for (int off = 64; off > 0; off >>= 1) {
    if (t < off) red[t] += red[t + off];
    __syncthreads();
  }
  if (t == 0) pred[b * 18 + col] = red[0] + bt[col];
}

// ---------------- theta from consecutive point pairs ----------------
__global__ __launch_bounds__(256)
void k_theta(const float* __restrict__ pred, float* __restrict__ theta, int n) {
  const int t = blockIdx.x * blockDim.x + threadIdx.x;
  if (t >= 32 * n) return;
  const int b = t / n, j = t % n, m = n + 1;
  const float* p = pred + (size_t)(b * m + j) * 2;
  const float x0 = p[0], y0 = p[1], x1 = p[2], y1 = p[3];
  const float cx = (x0 + x1) * 0.5f, cy = (y0 + y1) * 0.5f;
  const float dx = (x1 - x0) * 0.5f, dy = (y1 - y0) * 0.5f;
  float* th = theta + (size_t)t * 6;
  th[0] = dx; th[1] = -dy; th[2] = -cx;
  th[3] = dy; th[4] = dx;  th[5] = -cy;
}

// ---------------- bilinear ROI sampler -> f16 activation matrix (M x 49C) ----------------
__global__ void k_sample(const float* __restrict__ feat, const float* __restrict__ theta,
                         _Float16* __restrict__ X, int n, int H, int W, int C) {
  const int pid = blockIdx.x;          // (b*n + j)*49 + s
  const int s = pid % 49, roi = pid / 49;
  const int iy = s / 7, ix = s % 7;
  const float* th = theta + (size_t)roi * 6;
  const float gx = (float)(ix - 3) * (1.f / 3.f);
  const float gy = (float)(iy - 3) * (1.f / 3.f);
  const float xs = th[0] * gx + th[1] * gy + th[2];
  const float ys = th[3] * gx + th[4] * gy + th[5];
  const float px = (xs + 1.f) * (float)(W - 1) * 0.5f;
  const float py = (ys + 1.f) * (float)(H - 1) * 0.5f;
  const float x0 = floorf(px), y0 = floorf(py);
  const float wx = px - x0, wy = py - y0;
  const int x0i = (int)fminf(fmaxf(x0, 0.f), (float)(W - 1));
  const int x1i = (int)fminf(fmaxf(x0 + 1.f, 0.f), (float)(W - 1));
  const int y0i = (int)fminf(fmaxf(y0, 0.f), (float)(H - 1));
  const int y1i = (int)fminf(fmaxf(y0 + 1.f, 0.f), (float)(H - 1));
  const float wa = (1.f - wx) * (1.f - wy), wb = (1.f - wx) * wy;
  const float wc = wx * (1.f - wy),         wd = wx * wy;
  const int b = roi / n;
  const size_t base = (size_t)b * H * W * C;
  const float* Ia = feat + base + ((size_t)y0i * W + x0i) * C;
  const float* Ib = feat + base + ((size_t)y1i * W + x0i) * C;
  const float* Ic = feat + base + ((size_t)y0i * W + x1i) * C;
  const float* Id = feat + base + ((size_t)y1i * W + x1i) * C;
  _Float16* dst = X + (size_t)roi * 49 * C + (size_t)s * C;
  for (int c = threadIdx.x; c < C; c += blockDim.x)
    dst[c] = (_Float16)(wa * Ia[c] + wb * Ib[c] + wc * Ic[c] + wd * Id[c]);
}

// ---------------- WMMA GEMM: Out = tanh(A(f16) @ W(f32->f16) + bias), f16 out ----------------
// A: M x K row-major f16 (M % 64 == 0, K % 32 == 0), W: K x N row-major f32 (N % 64 == 0)
__global__ __launch_bounds__(256)
void k_gemm_tanh(const _Float16* __restrict__ A, const float* __restrict__ W,
                 const float* __restrict__ bias, _Float16* __restrict__ Out,
                 int N, int K) {
  __shared__ _Float16 sA[BM * LDT];
  __shared__ _Float16 sB[BN * LDT];
  const int tid  = threadIdx.x;
  const int lane = tid & 31;
  const int wave = tid >> 5;        // 8 waves
  const int wm   = wave & 3;        // 4 row tiles of 16
  const int wn   = wave >> 2;       // 2 col tiles of 32
  const int bm   = blockIdx.x * BM;
  const int bn   = blockIdx.y * BN;

  v8f acc0 = {};
  v8f acc1 = {};

  const int ar  = tid >> 2;         // A stage: row 0..63
  const int ac  = (tid & 3) * 8;    //          col 0,8,16,24
  const int bkr = tid >> 3;         // B stage: k 0..31
  const int bn0 = (tid & 7) * 8;    //          n 0..56

  const _Float16* Aptr = A + (size_t)(bm + ar) * K + ac;
  const float*    Wptr = W + (size_t)bkr * N + bn + bn0;

  const int m  = lane & 15;
  const int hi = lane >> 4;

  for (int k0 = 0; k0 < K; k0 += BK) {
    // stage A tile (row-major): async DMA global->LDS on gfx1250 (ASYNCcnt path)
#if HAVE_ASYNC_LDS
    __builtin_amdgcn_global_load_async_to_lds_b128(
        (g_v4i*)(Aptr + k0), (l_v4i*)&sA[ar * LDT + ac], 0, 0);
#else
    *(uint4*)&sA[ar * LDT + ac] = *(const uint4*)(Aptr + k0);
#endif
    // stage B tile: fp32 -> f16, transposed to [n][k] (needs VALU convert -> synchronous)
    const float4 w0 = *(const float4*)(Wptr + (size_t)k0 * N);
    const float4 w1 = *(const float4*)(Wptr + (size_t)k0 * N + 4);
    const _Float16 hv[8] = { (_Float16)w0.x, (_Float16)w0.y, (_Float16)w0.z, (_Float16)w0.w,
                             (_Float16)w1.x, (_Float16)w1.y, (_Float16)w1.z, (_Float16)w1.w };
#pragma unroll
    for (int i = 0; i < 8; ++i) sB[(bn0 + i) * LDT + bkr] = hv[i];

    wait_asynccnt0();               // barrier does NOT cover ASYNCcnt
    __syncthreads();

    if (k0 + BK < K) {                      // gfx1250 global_prefetch_b8
      __builtin_prefetch(Aptr + k0 + BK, 0, 1);
      __builtin_prefetch(Wptr + (size_t)(k0 + BK) * N, 0, 1);
    }

    // A fragment: lane<16 -> K 0-15; lane>=16 -> K 16-31 of row m
    const int abase = (wm * 16 + m) * LDT + hi * 16;
    const v8h alo = *(const v8h*)&sA[abase];
    const v8h ahi = *(const v8h*)&sA[abase + 8];
    const v16h va = __builtin_shufflevector(alo, ahi, 0,1,2,3,4,5,6,7,8,9,10,11,12,13,14,15);

    // B fragments: lane<16 -> K 0-15, lane>=16 -> K 16-31 of column m (+16)
    const int bb0 = (wn * 32 + m) * LDT + hi * 16;
    const v8h b0l = *(const v8h*)&sB[bb0];
    const v8h b0h = *(const v8h*)&sB[bb0 + 8];
    const v16h vb0 = __builtin_shufflevector(b0l, b0h, 0,1,2,3,4,5,6,7,8,9,10,11,12,13,14,15);
    const int bb1 = (wn * 32 + 16 + m) * LDT + hi * 16;
    const v8h b1l = *(const v8h*)&sB[bb1];
    const v8h b1h = *(const v8h*)&sB[bb1 + 8];
    const v16h vb1 = __builtin_shufflevector(b1l, b1h, 0,1,2,3,4,5,6,7,8,9,10,11,12,13,14,15);

    acc0 = __builtin_amdgcn_wmma_f32_16x16x32_f16(false, va, false, vb0, (short)0, acc0, false, false);
    acc1 = __builtin_amdgcn_wmma_f32_16x16x32_f16(false, va, false, vb1, (short)0, acc1, false, false);
    __syncthreads();
  }

  // epilogue: C layout -> row = r + 8*hi, col = lane&15
  const int col0 = bn + wn * 32 + m;
  const int col1 = col0 + 16;
  const float bia0 = bias[col0], bia1 = bias[col1];
#pragma unroll
  for (int r = 0; r < 8; ++r) {
    const int row = bm + wm * 16 + hi * 8 + r;
    Out[(size_t)row * N + col0] = (_Float16)tanhf(acc0[r] + bia0);
    Out[(size_t)row * N + col1] = (_Float16)tanhf(acc1[r] + bia1);
  }
}

// ---------------- f3 = f2 @ W3 + b3 + const; then affine transform to pt ----------------
__global__ __launch_bounds__(256)
void k_f3pt(const _Float16* __restrict__ f2, const float* __restrict__ W3,
            const float* __restrict__ b3, const float* __restrict__ theta,
            float* __restrict__ pt, int n) {
  const int rr = blockIdx.x * blockDim.x + threadIdx.x;
  if (rr >= 32 * n) return;
  float o[6] = {0.f, 0.f, 0.f, 0.f, 0.f, 0.f};
  const _Float16* f = f2 + (size_t)rr * 256;
  for (int k = 0; k < 256; ++k) {
    const float v = (float)f[k];
    const float* w = W3 + k * 6;
#pragma unroll
    for (int i = 0; i < 6; ++i) o[i] += v * w[i];
  }
  const float cadd[6] = {-1.f, 0.f, 0.f, 0.f, 1.f, 0.f};
#pragma unroll
  for (int i = 0; i < 6; ++i) o[i] += b3[i] + cadd[i];
  const float* th = theta + (size_t)rr * 6;
  const int b = rr / n, j = rr % n;
  float* P = pt + ((size_t)b * (3 * n) + 3 * j) * 2;
#pragma unroll
  for (int q = 0; q < 3; ++q) {
    const float px = o[2 * q], py = o[2 * q + 1];
    P[2 * q + 0] = th[0] * px + th[1] * py - th[2];   // hom z = -1
    P[2 * q + 1] = th[3] * px + th[4] * py - th[5];
  }
}

// ---------------- gather_pairs + mean -> new pred (B x (2n+1) x 2) ----------------
__global__ __launch_bounds__(256)
void k_gather(const float* __restrict__ pt, float* __restrict__ predo, int n) {
  const int P = 2 * n + 1;
  const int t = blockIdx.x * blockDim.x + threadIdx.x;
  if (t >= 32 * P) return;
  const int b = t / P, k = t % P;
  const int niter = n - 1;
  int i0, i1;
  if (k == 0) { i0 = i1 = 0; }
  else if (k <= 2 * niter) {
    const int j = (k - 1) >> 1;
    if (k & 1) { i0 = i1 = 1 + 3 * j; }
    else       { i0 = 2 + 3 * j; i1 = 3 + 3 * j; }
  } else if (k == 2 * niter + 1) { i0 = i1 = 3 * n - 2; }
  else { i0 = i1 = 3 * n - 1; }
  const float* p0 = pt + ((size_t)b * 3 * n + i0) * 2;
  const float* p1 = pt + ((size_t)b * 3 * n + i1) * 2;
  float* o = predo + ((size_t)b * P + k) * 2;
  o[0] = 0.5f * (p0[0] + p1[0]);
  o[1] = 0.5f * (p0[1] + p1[1]);
}

// ---------------- final: out = pred[:, :64, :] * 6 ----------------
__global__ __launch_bounds__(256)
void k_out(const float* __restrict__ pred, float* __restrict__ out) {
  const int t = blockIdx.x * blockDim.x + threadIdx.x;
  if (t >= 32 * 64 * 2) return;
  const int b = t / 128, rem = t % 128;
  out[t] = pred[b * 65 * 2 + rem] * 6.f;
}

extern "C" void kernel_launch(void* const* d_in, const int* in_sizes, int n_in,
                              void* d_out, int out_size, void* d_ws, size_t ws_size,
                              hipStream_t stream) {
  (void)in_sizes; (void)n_in; (void)out_size; (void)ws_size;
  const float* last_feature = (const float*)d_in[0];
  const float* feats[3] = { (const float*)d_in[1], (const float*)d_in[2], (const float*)d_in[3] };
  const float* W_trans = (const float*)d_in[4];
  const float* b_trans = (const float*)d_in[5];

  char* ws = (char*)d_ws;
  size_t off = 0;
  auto carve = [&](size_t bytes) { char* p = ws + off; off += (bytes + 255) & ~(size_t)255; return p; };
  float*    predA = (float*)carve((size_t)32 * 65 * 2 * 4);
  float*    predB = (float*)carve((size_t)32 * 65 * 2 * 4);
  float*    theta = (float*)carve((size_t)32 * 32 * 6 * 4);
  float*    pt    = (float*)carve((size_t)32 * 96 * 2 * 4);
  _Float16* X     = (_Float16*)carve((size_t)6422528 * 2);     // M*K == 6,422,528 for every head
  _Float16* f1    = (_Float16*)carve((size_t)1024 * 1024 * 2);
  _Float16* f2    = (_Float16*)carve((size_t)1024 * 256 * 2);

  k_pred0<<<dim3(32, 18), 128, 0, stream>>>(last_feature, W_trans, b_trans, predA);

  float* cur = predA;
  float* nxt = predB;
  for (int h = 0; h < 3; ++h) {
    const int n  = 8 << h;            // 8, 16, 32 thetas
    const int C  = 512 >> h;          // 512, 256, 128 channels
    const int HW = 28 << h;           // 28, 56, 112
    const int K  = 49 * C;            // 25088, 12544, 6272 (all % 32 == 0)
    const int M  = 32 * n;            // 256, 512, 1024   (all % 64 == 0)
    const float* W1 = (const float*)d_in[6 + 6 * h + 0];
    const float* b1 = (const float*)d_in[6 + 6 * h + 1];
    const float* W2 = (const float*)d_in[6 + 6 * h + 2];
    const float* b2 = (const float*)d_in[6 + 6 * h + 3];
    const float* W3 = (const float*)d_in[6 + 6 * h + 4];
    const float* b3 = (const float*)d_in[6 + 6 * h + 5];

    const int nt = 32 * n;
    k_theta<<<(nt + 255) / 256, 256, 0, stream>>>(cur, theta, n);
    k_sample<<<dim3(32 * n * 49), dim3(C < 256 ? C : 256), 0, stream>>>(feats[h], theta, X, n, HW, HW, C);
    k_gemm_tanh<<<dim3(M / BM, 1024 / BN), 256, 0, stream>>>(X, W1, b1, f1, 1024, K);
    k_gemm_tanh<<<dim3(M / BM, 256 / BN), 256, 0, stream>>>(f1, W2, b2, f2, 256, 1024);
    k_f3pt<<<(M + 255) / 256, 256, 0, stream>>>(f2, W3, b3, theta, pt, n);
    const int P = 2 * n + 1;
    k_gather<<<(32 * P + 255) / 256, 256, 0, stream>>>(pt, nxt, n);
    float* tmp = cur; cur = nxt; nxt = tmp;
  }

  k_out<<<(4096 + 255) / 256, 256, 0, stream>>>(cur, (float*)d_out);
}
